// RoutedQwenAttention_54692113547530
// MI455X (gfx1250) — matrix operand order repaired and checked by reading, not compiled, not run on hardware
//
#include <hip/hip_runtime.h>
#include <math.h>

// ---------------- types ----------------
typedef __bf16 bf16;
typedef bf16  bf16x16 __attribute__((ext_vector_type(16)));
typedef float f32x8   __attribute__((ext_vector_type(8)));
typedef int   i32x4   __attribute__((ext_vector_type(4)));

// problem constants (fixed shapes)
#define CB   2
#define CS   2048
#define CE   2048
#define CH   16
#define CDH  128
#define CM   (CB*CS)     // 4096 rows
#define CN   (CH*CDH)    // 2048 out features
#define CRL  32          // LoRA rank
#define CNX  64          // experts
#define CSCALE 1.0f

// ---------------- WMMA helpers ----------------
__device__ __forceinline__ f32x8 wmma_bf16(bf16x16 a, bf16x16 b, f32x8 c) {
  // v_wmma_f32_16x16x32_bf16  D = A(16x32) * B(32x16) + C
  return __builtin_amdgcn_wmma_f32_16x16x32_bf16(
      false, a, false, b, (short)0, c, false, false);
}

// A fragment (16x32, bf16), source row-major [M,K], 16B loads.
// lane L<16: row m0+L, K = k0+{0..7, 16..23}; lane>=16: row m0+L-16, K = k0+{8..15, 24..31}
__device__ __forceinline__ bf16x16 load_a_frag(const bf16* p, int row0, int k0, int ld) {
  int lane = (int)(threadIdx.x & 31u);
  const bf16* r = p + (size_t)(row0 + (lane & 15)) * (size_t)ld + k0 + ((lane >> 4) << 3);
  union { struct { i32x4 lo; i32x4 hi; } q; bf16x16 v; } u;
  u.q.lo = *(const i32x4*)(r);
  u.q.hi = *(const i32x4*)(r + 16);
  return u.v;
}

// B fragment (32x16, bf16), source row-major [N,K] (each column n is a contiguous K row).
// lane L<16: col n0+L, K = k0+{0..7, 16..23}; lane>=16: col n0+L-16, K = k0+{8..15, 24..31}
__device__ __forceinline__ bf16x16 load_b_frag(const bf16* p, int k0, int col0, int ld) {
  int lane = (int)(threadIdx.x & 31u);
  const bf16* r = p + (size_t)(col0 + (lane & 15)) * (size_t)ld + k0 + ((lane >> 4) << 3);
  union { struct { i32x4 lo; i32x4 hi; } q; bf16x16 v; } u;
  u.q.lo = *(const i32x4*)(r);
  u.q.hi = *(const i32x4*)(r + 16);
  return u.v;
}

__device__ __forceinline__ f32x8 zero8() {
  f32x8 z = {0.f,0.f,0.f,0.f,0.f,0.f,0.f,0.f};
  return z;
}
__device__ __forceinline__ float siluf(float v) { return v / (1.0f + __expf(-v)); }

// half-wave (16-lane) reductions: rows of a C fragment live in 16 lanes
__device__ __forceinline__ float hmax16(float x) {
  #pragma unroll
  for (int m = 1; m < 16; m <<= 1) x = fmaxf(x, __shfl_xor(x, m, 32));
  return x;
}
__device__ __forceinline__ float hsum16(float x) {
  #pragma unroll
  for (int m = 1; m < 16; m <<= 1) x += __shfl_xor(x, m, 32);
  return x;
}

// CDNA5 async copy: LDS[lds_off] = MEM[gaddr], 16B per lane, ASYNCcnt-tracked
__device__ __forceinline__ void async_copy16(uint32_t lds_off, const void* gaddr) {
  asm volatile("global_load_async_to_lds_b128 %0, %1, off"
               :: "v"(lds_off), "v"(gaddr) : "memory");
}
__device__ __forceinline__ void wait_async0() {
  asm volatile("s_wait_asynccnt 0x0" ::: "memory");
}

// ---------------- kernel: fp32 -> bf16 convert ----------------
__global__ __launch_bounds__(256) void cvt_kernel(const float* __restrict__ in,
                                                  bf16* __restrict__ out, int n) {
  int i = (int)(blockIdx.x * 256u + threadIdx.x);
  if (i < n) out[i] = (bf16)in[i];
}

// ---------------- kernel: out = silu(x @ Wt) [* scale] [* mask], bf16 out ----------------
template<int NT, bool MASK>
__global__ __launch_bounds__(256) void actgemm_kernel(const bf16* __restrict__ xb,
                                                      const bf16* __restrict__ wb,
                                                      const float* __restrict__ mask,
                                                      bf16* __restrict__ out,
                                                      float scale) {
  constexpr int K = CE;
  constexpr int NF = NT / 16;
  int w  = (int)(threadIdx.x >> 5);
  int lane = (int)(threadIdx.x & 31u), ln = lane & 15, hl = lane >> 4;
  int m0 = (int)blockIdx.x * 128 + w * 16;

  f32x8 acc[NF];
  #pragma unroll
  for (int nf = 0; nf < NF; ++nf) acc[nf] = zero8();

  for (int k = 0; k < K; k += 32) {
    bf16x16 a = load_a_frag(xb, m0, k, K);
    #pragma unroll
    for (int nf = 0; nf < NF; ++nf) {
      bf16x16 b = load_b_frag(wb, k, nf * 16, K);
      acc[nf] = wmma_bf16(a, b, acc[nf]);
    }
  }
  #pragma unroll
  for (int nf = 0; nf < NF; ++nf) {
    #pragma unroll
    for (int r = 0; r < 8; ++r) {
      int m = m0 + r + 8 * hl;
      int n = nf * 16 + ln;
      float v = siluf(acc[nf][r]) * scale;
      if (MASK) v *= mask[(size_t)m * NT + n];
      out[(size_t)m * NT + n] = (bf16)v;
    }
  }
}

// ---------------- kernel: fused routed projection ----------------
// out = x@Wt + bias + (h @ Bt) + (g @ Rt)
// wave tile 32x64 (2x4 WMMA frags), block 128x128 (8 waves = 4m x 2n)
// MODE 0: bf16 row-major [CM,CN]; MODE 1: bf16 V-transposed [B,H,Dh,S]; MODE 2: fp32 [CM,CN]
template<int MODE>
__global__ __launch_bounds__(256) void routed_proj_kernel(const bf16* __restrict__ xb,
                                                          const bf16* __restrict__ Wb,
                                                          const float* __restrict__ bias,
                                                          const bf16* __restrict__ hb,
                                                          const bf16* __restrict__ Bb,
                                                          const bf16* __restrict__ gb,
                                                          const bf16* __restrict__ Rb,
                                                          void* __restrict__ outp) {
  constexpr int K = CE;
  int w  = (int)(threadIdx.x >> 5);
  int lane = (int)(threadIdx.x & 31u), ln = lane & 15, hl = lane >> 4;
  int m0 = (int)blockIdx.y * 128 + (w >> 1) * 32;
  int n0 = (int)blockIdx.x * 128 + (w & 1) * 64;

  f32x8 acc[2][4];
  #pragma unroll
  for (int i = 0; i < 2; ++i)
    #pragma unroll
    for (int j = 0; j < 4; ++j) acc[i][j] = zero8();

  // dense path: K = 2048
  for (int k = 0; k < K; k += 32) {
    if (k + 64 < K) {
      __builtin_prefetch((const void*)(xb + (size_t)(m0 + ln) * K + k + 64), 0, 3);
      __builtin_prefetch((const void*)(Wb + (size_t)(n0 + ln) * K + k + 64), 0, 3);
      __builtin_prefetch((const void*)(Wb + (size_t)(n0 + 32 + ln) * K + k + 64), 0, 3);
    }
    bf16x16 a0 = load_a_frag(xb, m0,      k, K);
    bf16x16 a1 = load_a_frag(xb, m0 + 16, k, K);
    #pragma unroll
    for (int j = 0; j < 4; ++j) {
      bf16x16 b = load_b_frag(Wb, k, n0 + j * 16, K);
      acc[0][j] = wmma_bf16(a0, b, acc[0][j]);
      acc[1][j] = wmma_bf16(a1, b, acc[1][j]);
    }
  }
  // LoRA path: K = 32 (h already has silu*scale applied)
  {
    bf16x16 a0 = load_a_frag(hb, m0,      0, CRL);
    bf16x16 a1 = load_a_frag(hb, m0 + 16, 0, CRL);
    #pragma unroll
    for (int j = 0; j < 4; ++j) {
      bf16x16 b = load_b_frag(Bb, 0, n0 + j * 16, CRL);
      acc[0][j] = wmma_bf16(a0, b, acc[0][j]);
      acc[1][j] = wmma_bf16(a1, b, acc[1][j]);
    }
  }
  // routed path: K = 64 (g already has silu*mask applied)
  #pragma unroll
  for (int k = 0; k < CNX; k += 32) {
    bf16x16 a0 = load_a_frag(gb, m0,      k, CNX);
    bf16x16 a1 = load_a_frag(gb, m0 + 16, k, CNX);
    #pragma unroll
    for (int j = 0; j < 4; ++j) {
      bf16x16 b = load_b_frag(Rb, k, n0 + j * 16, CNX);
      acc[0][j] = wmma_bf16(a0, b, acc[0][j]);
      acc[1][j] = wmma_bf16(a1, b, acc[1][j]);
    }
  }

  // epilogue
  #pragma unroll
  for (int i = 0; i < 2; ++i) {
    #pragma unroll
    for (int j = 0; j < 4; ++j) {
      #pragma unroll
      for (int r = 0; r < 8; ++r) {
        int m = m0 + 16 * i + r + 8 * hl;
        int n = n0 + 16 * j + ln;
        float v = acc[i][j][r];
        if (bias) v += bias[n];
        if (MODE == 0) {
          ((bf16*)outp)[(size_t)m * CN + n] = (bf16)v;
        } else if (MODE == 1) {
          int b = m >> 11, s = m & (CS - 1);
          int hh = n >> 7, d = n & (CDH - 1);
          ((bf16*)outp)[(((size_t)b * CH + hh) * CDH + d) * CS + s] = (bf16)v;
        } else {
          ((float*)outp)[(size_t)m * CN + n] = v;
        }
      }
    }
  }
}

// ---------------- kernel: in-place RoPE on bf16 [B,S,H,Dh] ----------------
__global__ __launch_bounds__(256) void rope_kernel(bf16* __restrict__ t) {
  size_t i = (size_t)blockIdx.x * 256 + threadIdx.x;     // one thread per (b,s,h,d<64) pair
  if (i >= (size_t)CB * CS * CH * (CDH / 2)) return;
  int d = (int)(i & 63); size_t rest = i >> 6;
  int h = (int)(rest & 15); rest >>= 4;
  int s = (int)(rest & (CS - 1)); int b = (int)(rest >> 11);
  size_t base = (((size_t)(b * CS + s)) * CH + h) * CDH;
  float x0 = (float)t[base + d];
  float x1 = (float)t[base + d + 64];
  float inv = __powf(10000.0f, -(float)d * (1.0f / 64.0f));
  float ang = (float)s * inv;
  float c = __cosf(ang), sn = __sinf(ang);
  t[base + d]      = (bf16)(x0 * c - x1 * sn);
  t[base + d + 64] = (bf16)(x1 * c + x0 * sn);
}

// ---------------- kernel: causal flash attention, async-LDS staged K/V ----------------
// qb,kb: bf16 [B,S,H,Dh] (post-RoPE); vt: bf16 [B,H,Dh,S]; ob: bf16 [B,S,H*Dh]
// Block = 4 waves x 16 query rows = 64 queries. K/V tiles (32 keys) staged to LDS with
// GLOBAL_LOAD_ASYNC_TO_LDS_B128 (ASYNCcnt), double-buffered across the key-tile loop.
__global__ __launch_bounds__(128) void flash_attn_kernel(const bf16* __restrict__ qb,
                                                         const bf16* __restrict__ kb,
                                                         const bf16* __restrict__ vt,
                                                         bf16* __restrict__ ob) {
  __shared__ bf16 Kt[2][32][128];                  // [buf][key][d]  2 x 8KB
  __shared__ bf16 Vt[2][128][32];                  // [buf][d][key]  2 x 8KB
  __shared__ bf16 plds[4][16][32];                 // P tile per wave (C-layout -> A-layout)

  int w = (int)(threadIdx.x >> 5);
  int lane = (int)(threadIdx.x & 31u), ln = lane & 15, hl = lane >> 4;
  int q0 = (int)blockIdx.x * 64 + w * 16;          // 16 query rows per wave
  int bh = (int)blockIdx.y; int b = bh >> 4; int h = bh & 15;

  const bf16* qbase = qb + (((size_t)(b * CS + q0)) * CH + h) * CDH;  // row stride H*Dh
  const bf16* kbase = kb + ((size_t)b * CS * CH + h) * CDH;           // key stride H*Dh
  const bf16* vbase = vt + (((size_t)(b * CH + h)) * CDH) * CS;       // [Dh][S]

  // issue async copies of key-tile kt into LDS buffer buf (whole block cooperates)
  auto stage_tile = [&](int kt, int buf) {
    #pragma unroll
    for (int i = 0; i < 4; ++i) {
      // K tile: 32 rows x 256B = 512 x 16B chunks
      int ck = (int)threadIdx.x + i * 128;
      int krow = ck >> 4, kofs = (ck & 15) << 4;   // byte offset within row
      const bf16* gk = kbase + (size_t)(kt * 32 + krow) * (CH * CDH) + (kofs >> 1);
      async_copy16((uint32_t)(size_t)(&Kt[buf][krow][0]) + (uint32_t)kofs, gk);
      // V tile: 128 rows x 64B = 512 x 16B chunks
      int vrow = ck >> 2, vofs = (ck & 3) << 4;
      const bf16* gv = vbase + (size_t)vrow * CS + kt * 32 + (vofs >> 1);
      async_copy16((uint32_t)(size_t)(&Vt[buf][vrow][0]) + (uint32_t)vofs, gv);
    }
  };

  bf16x16 qa[4];
  #pragma unroll
  for (int kk = 0; kk < 4; ++kk) qa[kk] = load_a_frag(qbase, 0, kk * 32, CH * CDH);

  f32x8 oacc[8];
  #pragma unroll
  for (int nf = 0; nf < 8; ++nf) oacc[nf] = zero8();
  float mrow[8], lrow[8];
  #pragma unroll
  for (int r = 0; r < 8; ++r) { mrow[r] = -1e30f; lrow[r] = 0.f; }

  const float scl = 0.08838834764831845f;          // 1/sqrt(128)
  int nkt_w   = q0 / 32 + 1;                       // this wave's causal bound
  int nkt_blk = (int)blockIdx.x * 2 + 2;           // block-uniform tile count

  stage_tile(0, 0);
  wait_async0();
  __syncthreads();

  for (int kt = 0; kt < nkt_blk; ++kt) {
    int buf = kt & 1;
    if (kt + 1 < nkt_blk) stage_tile(kt + 1, buf ^ 1);   // prefetch next tile (async)

    if (kt < nkt_w) {
      // scores S = Q @ K^T from LDS K tile
      f32x8 s0 = zero8(), s1 = zero8();
      #pragma unroll
      for (int kk = 0; kk < 4; ++kk) {
        bf16x16 b0 = load_b_frag(&Kt[buf][0][0], kk * 32, 0,  CDH);
        bf16x16 b1 = load_b_frag(&Kt[buf][0][0], kk * 32, 16, CDH);
        s0 = wmma_bf16(qa[kk], b0, s0);
        s1 = wmma_bf16(qa[kk], b1, s1);
      }
      // online softmax update per owned row element
      #pragma unroll
      for (int r = 0; r < 8; ++r) {
        int qrow = q0 + r + 8 * hl;
        float v0 = s0[r] * scl, v1 = s1[r] * scl;
        if (kt * 32 + ln      > qrow) v0 = -1e30f;
        if (kt * 32 + 16 + ln > qrow) v1 = -1e30f;
        float mx   = hmax16(fmaxf(v0, v1));
        float mnew = fmaxf(mrow[r], mx);
        float corr = __expf(mrow[r] - mnew);
        float p0 = __expf(v0 - mnew), p1 = __expf(v1 - mnew);
        lrow[r] = lrow[r] * corr + hsum16(p0 + p1);
        mrow[r] = mnew;
        #pragma unroll
        for (int nf = 0; nf < 8; ++nf) oacc[nf][r] *= corr;
        plds[w][r + 8 * hl][ln]      = (bf16)p0;   // wave-private; DScnt ordered
        plds[w][r + 8 * hl][ln + 16] = (bf16)p1;
      }
      // O += P @ V  (K = 32 keys, N = 128 dims) from LDS V tile
      bf16x16 pa = load_a_frag(&plds[w][0][0], 0, 0, 32);
      #pragma unroll
      for (int nf = 0; nf < 8; ++nf) {
        bf16x16 vb = load_b_frag(&Vt[buf][0][0], 0, nf * 16, 32);
        oacc[nf] = wmma_bf16(pa, vb, oacc[nf]);
      }
    }
    wait_async0();          // next tile resident before anyone reads it
    __syncthreads();        // and everyone done reading current before overwrite
  }

  float invl[8];
  #pragma unroll
  for (int r = 0; r < 8; ++r) invl[r] = 1.0f / lrow[r];
  size_t orow0 = (size_t)(b * CS + q0);
  #pragma unroll
  for (int nf = 0; nf < 8; ++nf) {
    #pragma unroll
    for (int r = 0; r < 8; ++r) {
      ob[(orow0 + r + 8 * hl) * (size_t)CN + h * CDH + nf * 16 + ln] =
          (bf16)(oacc[nf][r] * invl[r]);
    }
  }
}

// ---------------- host orchestration ----------------
extern "C" void kernel_launch(void* const* d_in, const int* in_sizes, int n_in,
                              void* d_out, int out_size, void* d_ws, size_t ws_size,
                              hipStream_t stream) {
  (void)in_sizes; (void)n_in; (void)out_size; (void)ws_size;

  const float* x    = (const float*)d_in[0];
  const float* mask = (const float*)d_in[1];
  const float *Wf[4], *bf_[3], *Af[4], *Bf[4], *Cf[4], *Rf[4];
  for (int p = 0; p < 3; ++p) {
    Wf[p]  = (const float*)d_in[2 + p * 6 + 0];
    bf_[p] = (const float*)d_in[2 + p * 6 + 1];
    Af[p]  = (const float*)d_in[2 + p * 6 + 2];
    Bf[p]  = (const float*)d_in[2 + p * 6 + 3];
    Cf[p]  = (const float*)d_in[2 + p * 6 + 4];
    Rf[p]  = (const float*)d_in[2 + p * 6 + 5];
  }
  Wf[3] = (const float*)d_in[20];
  Af[3] = (const float*)d_in[21];
  Bf[3] = (const float*)d_in[22];
  Cf[3] = (const float*)d_in[23];
  Rf[3] = (const float*)d_in[24];

  // carve workspace (all bf16 staging)
  size_t off = 0;
  auto carve = [&](size_t elems) -> bf16* {
    bf16* p = (bf16*)((char*)d_ws + off);
    off += ((elems * sizeof(bf16)) + 255) & ~(size_t)255;
    return p;
  };
  bf16* xb = carve((size_t)CM * CE);
  bf16 *Wb[4], *Ab[4], *Bb[4], *Cb[4], *Rb[4];
  for (int p = 0; p < 4; ++p) {
    Wb[p] = carve((size_t)CN * CE);
    Ab[p] = carve((size_t)CRL * CE);
    Bb[p] = carve((size_t)CN * CRL);
    Cb[p] = carve((size_t)CNX * CE);
    Rb[p] = carve((size_t)CN * CNX);
  }
  bf16* hbuf = carve((size_t)CM * CRL);
  bf16* gbuf = carve((size_t)CM * CNX);
  bf16* qbuf = carve((size_t)CM * CN);
  bf16* kbuf = carve((size_t)CM * CN);
  bf16* vtbf = carve((size_t)CM * CN);   // [B,H,Dh,S]
  bf16* obuf = carve((size_t)CM * CN);

  auto cvt = [&](const float* src, bf16* dst, int n) {
    cvt_kernel<<<dim3((n + 255) / 256), dim3(256), 0, stream>>>(src, dst, n);
  };

  // stage 0: convert everything to bf16 (weights all land in [N,K] = B-operand layout)
  cvt(x, xb, CM * CE);
  for (int p = 0; p < 4; ++p) {
    cvt(Wf[p], Wb[p], CN * CE);
    cvt(Af[p], Ab[p], CRL * CE);
    cvt(Bf[p], Bb[p], CN * CRL);
    cvt(Cf[p], Cb[p], CNX * CE);
    cvt(Rf[p], Rb[p], CN * CNX);
  }

  dim3 gAct(CM / 128), bAct(256);
  dim3 gProj(CN / 128, CM / 128), bProj(256);

  // stage 1: Q/K/V routed projections
  void* proj_out[3] = { (void*)qbuf, (void*)kbuf, (void*)vtbf };
  for (int p = 0; p < 3; ++p) {
    actgemm_kernel<CRL, false><<<gAct, bAct, 0, stream>>>(xb, Ab[p], nullptr, hbuf, CSCALE);
    actgemm_kernel<CNX, true ><<<gAct, bAct, 0, stream>>>(xb, Cb[p], mask,    gbuf, 1.0f);
    if (p < 2)
      routed_proj_kernel<0><<<gProj, bProj, 0, stream>>>(xb, Wb[p], bf_[p], hbuf, Bb[p], gbuf, Rb[p], proj_out[p]);
    else
      routed_proj_kernel<1><<<gProj, bProj, 0, stream>>>(xb, Wb[p], bf_[p], hbuf, Bb[p], gbuf, Rb[p], proj_out[p]);
  }

  // stage 2: RoPE on q and k (in place, bf16)
  {
    size_t n = (size_t)CB * CS * CH * (CDH / 2);
    dim3 g((unsigned)((n + 255) / 256));
    rope_kernel<<<g, dim3(256), 0, stream>>>(qbuf);
    rope_kernel<<<g, dim3(256), 0, stream>>>(kbuf);
  }

  // stage 3: causal flash attention (async-LDS staged)
  flash_attn_kernel<<<dim3(CS / 64, CB * CH), dim3(128), 0, stream>>>(qbuf, kbuf, vtbf, obuf);

  // stage 4: output routed projection -> fp32 d_out
  actgemm_kernel<CRL, false><<<gAct, bAct, 0, stream>>>(obuf, Ab[3], nullptr, hbuf, CSCALE);
  actgemm_kernel<CNX, true ><<<gAct, bAct, 0, stream>>>(obuf, Cb[3], mask,    gbuf, 1.0f);
  routed_proj_kernel<2><<<gProj, bProj, 0, stream>>>(obuf, Wb[3], nullptr, hbuf, Bb[3], gbuf, Rb[3], d_out);
}
